// SelfAttentionMemory_7533372637187
// MI455X (gfx1250) — matrix operand order repaired and checked by reading, not compiled, not run on hardware
//
#include <hip/hip_runtime.h>
#include <hip/hip_bf16.h>

typedef __attribute__((ext_vector_type(16))) _Float16 v16h;
typedef __attribute__((ext_vector_type(8)))  _Float16 v8h;
typedef __attribute__((ext_vector_type(8)))  float    v8f;

#define BB 8
#define CC 64
#define NN 4096
#define QKD 8

__device__ __forceinline__ v8f wmma16(v16h a, v16h b, v8f c) {
    // D = A(16x32 f16) * B(32x16 f16) + C(16x16 f32)
    return __builtin_amdgcn_wmma_f32_16x16x32_f16(false, a, false, b, (short)0, c, false, false);
}

// ---------------------------------------------------------------------------
// Pack a weight matrix W[outC x inC] (fp32) into A-fragment layout, contiguous k order (as B):
// dst[((t*KC+kc)*32 + lane)*16 + h], lanes 0-15: M=t*16+lane, K = kc*32 + h
// lanes 16-31: M=t*16+lane-16, K = kc*32 + 16 + h
// ---------------------------------------------------------------------------
__global__ void pack_w(const float* __restrict__ W, _Float16* __restrict__ dst,
                       int outC, int inC, int T, int KC, float scale) {
    int total = T * KC * 32 * 16;
    int i = blockIdx.x * blockDim.x + threadIdx.x;
    if (i >= total) return;
    int h     = i & 15;
    int lane  = (i >> 4) & 31;
    int chunk = i >> 9;          // t*KC + kc
    int kc = chunk % KC;
    int t  = chunk / KC;
    int laneHi = lane >> 4;
    int row    = t * 16 + (lane & 15);
    int klocal = h + laneHi * 16;
    int K      = kc * 32 + klocal;
    float v = (row < outC && K < inC) ? W[row * inC + K] * scale : 0.0f;
    dst[i] = (_Float16)v;
}

// Pad an 8-element bias to 16 floats (rows 8..15 = 0), optional scale.
__global__ void pack_bias8(const float* __restrict__ b, float* __restrict__ dst, float scale) {
    int i = threadIdx.x;
    if (i < 16) dst[i] = (i < QKD) ? b[i] * scale : 0.0f;
}

// Transpose [B,C,N] fp32 -> [B,N,C] f16 for h and m simultaneously.
__global__ void transpose_f16(const float* __restrict__ H, const float* __restrict__ M,
                              _Float16* __restrict__ hT, _Float16* __restrict__ mT) {
    int i = blockIdx.x * blockDim.x + threadIdx.x;          // over B*C*N
    int n = i % NN;
    int c = (i / NN) % CC;
    int b = i / (NN * CC);
    size_t dst = ((size_t)(b * NN + n)) * CC + c;
    hT[dst] = (_Float16)H[i];
    mT[dst] = (_Float16)M[i];
}

// ---------------------------------------------------------------------------
// Projections: Q/Kh/Km rows [B][N][8] f16, Vh/Vm channel-major [B][C][N] f16.
// A = packed weights, B = activation columns from hT/mT.
// ---------------------------------------------------------------------------
__global__ __launch_bounds__(128) void proj_kernel(
    const _Float16* __restrict__ hT, const _Float16* __restrict__ mT,
    const _Float16* __restrict__ wqP,  const float* __restrict__ bqP,
    const _Float16* __restrict__ wkhP, const float* __restrict__ bkhP,
    const _Float16* __restrict__ wkmP, const float* __restrict__ bkmP,
    const _Float16* __restrict__ wvhP, const float* __restrict__ bvh,
    const _Float16* __restrict__ wvmP, const float* __restrict__ bvm,
    _Float16* __restrict__ qrow, _Float16* __restrict__ khrow, _Float16* __restrict__ kmrow,
    _Float16* __restrict__ vh,   _Float16* __restrict__ vm) {
    const int lane = threadIdx.x & 31;
    const int wave = threadIdx.x >> 5;
    const int b  = blockIdx.x / (NN / 64);
    const int n0 = (blockIdx.x % (NN / 64)) * 64 + wave * 16;
    const int laneHi = lane >> 4, laneLo = lane & 15;
    const int pos = n0 + laneLo;

    // B fragments: B[k=cin][n=pos]; lane holds column `pos`, halves = 16 K values.
    v16h Bh[2], Bm[2];
    for (int kc = 0; kc < 2; kc++) {
        Bh[kc] = *(const v16h*)&hT[((size_t)(b * NN + pos)) * CC + kc * 32 + 16 * laneHi];
        Bm[kc] = *(const v16h*)&mT[((size_t)(b * NN + pos)) * CC + kc * 32 + 16 * laneHi];
    }

    auto smallproj = [&](const _Float16* wp, const float* biasPad, const v16h* Bf,
                         _Float16* outrow) {
        v8f acc = *(const v8f*)&biasPad[8 * laneHi];          // one 32B load
        for (int kc = 0; kc < 2; kc++) {
            v16h a = *(const v16h*)&wp[((size_t)(kc * 32 + lane)) * 16];
            acc = wmma16(a, Bf[kc], acc);
        }
        if (laneHi == 0) {   // lanes 0-15 hold rows 0-7 = all 8 outputs for pos=lane
            v8h q;
            for (int v = 0; v < 8; v++) q[v] = (_Float16)acc[v];
            *(v8h*)&outrow[((size_t)(b * NN + n0 + laneLo)) * QKD] = q;
        }
    };
    smallproj(wqP,  bqP,  Bh, qrow);     // 1/sqrt(QK) folded into wq/bq at pack time
    smallproj(wkhP, bkhP, Bh, khrow);
    smallproj(wkmP, bkmP, Bm, kmrow);

    auto bigproj = [&](const _Float16* wp, const float* bias, const v16h* Bf,
                       _Float16* out) {
        for (int t = 0; t < 4; t++) {
            v8f acc = *(const v8f*)&bias[t * 16 + 8 * laneHi];
            for (int kc = 0; kc < 2; kc++) {
                v16h a = *(const v16h*)&wp[((size_t)((t * 2 + kc) * 32 + lane)) * 16];
                acc = wmma16(a, Bf[kc], acc);
            }
            for (int v = 0; v < 8; v++) {
                int c = t * 16 + v + 8 * laneHi;
                out[((size_t)(b * CC + c)) * NN + n0 + laneLo] = (_Float16)acc[v];
            }
        }
    };
    bigproj(wvhP, bvh, Bh, vh);
    bigproj(wvmP, bvm, Bm, vm);
}

// ---------------------------------------------------------------------------
// Fused dual flash-attention. One wave = one 16-query tile; 32-key blocks.
// Writes Zcat[b][n][0:64]=Zh, [64:128]=Zm as f16.
// ---------------------------------------------------------------------------
__global__ __launch_bounds__(128) void attn_kernel(
    const _Float16* __restrict__ qrow, const _Float16* __restrict__ khrow,
    const _Float16* __restrict__ kmrow, const _Float16* __restrict__ vh,
    const _Float16* __restrict__ vm, _Float16* __restrict__ zcat) {
    __shared__ alignas(32) _Float16 sKh[32 * 8];
    __shared__ alignas(32) _Float16 sKm[32 * 8];
    __shared__ alignas(32) _Float16 sVh[64 * 32];
    __shared__ alignas(32) _Float16 sVm[64 * 32];
    __shared__ alignas(32) _Float16 sP[4][2][16 * 32];

    const int tid = threadIdx.x, lane = tid & 31, wave = tid >> 5;
    const int b  = blockIdx.x / (NN / 64);
    const int q0 = (blockIdx.x % (NN / 64)) * 64 + wave * 16;
    const int laneHi = lane >> 4, laneLo = lane & 15;
    const _Float16 hmask = laneHi ? (_Float16)0.0f : (_Float16)1.0f;  // zero K-rows 8..31

    // Q A-fragment: rows = queries, K = 8 channels padded to 32 with zeros.
    // Branchless: unconditional load of a valid row, masked for lanes 16-31.
    v16h aQ = {};
    {
        v8h q = *(const v8h*)&qrow[((size_t)(b * NN + q0 + laneLo)) * QKD];
        q = q * hmask;                                      // v_pk_mul_f16, no branch
        for (int i = 0; i < 8; i++) aQ[i] = q[i];
    }

    float mh[8], lh[8], mmx[8], lm[8];
    v8f acch[4], accm[4];
    for (int v = 0; v < 8; v++) { mh[v] = -1e30f; lh[v] = 0.0f; mmx[v] = -1e30f; lm[v] = 0.0f; }
    for (int t = 0; t < 4; t++) { acch[t] = {}; accm[t] = {}; }

    for (int kb = 0; kb < NN / 32; kb++) {
        __syncthreads();
        // Stage K tiles (32 keys x 8 ch) and V tiles (64 ch x 32 keys) into LDS.
        if (tid < 32) {
            *(v8h*)&sKh[tid * 8] = *(const v8h*)&khrow[((size_t)(b * NN + kb * 32 + tid)) * QKD];
        } else if (tid < 64) {
            int k = tid - 32;
            *(v8h*)&sKm[k * 8] = *(const v8h*)&kmrow[((size_t)(b * NN + kb * 32 + k)) * QKD];
        }
        for (int rep = 0; rep < 4; rep++) {
            int chunk = tid + rep * 128;          // 0..511, 8 halves each
            int mat  = chunk >> 8;
            int cid  = chunk & 255;
            int row  = cid >> 2, part = cid & 3;
            const _Float16* src = mat ? vm : vh;
            _Float16*       dst = mat ? sVm : sVh;
            *(v8h*)&dst[row * 32 + part * 8] =
                *(const v8h*)&src[((size_t)(b * CC + row)) * NN + kb * 32 + part * 8];
        }
        __syncthreads();

        // Scores: S[16q x 16k] per sub-tile; channels 8..31 zero via hmask (branchless).
        v16h bk0 = {}, bk1 = {}, bm0 = {}, bm1 = {};
        {
            v8h t0 = *(const v8h*)&sKh[laneLo * 8]        * hmask;
            v8h t1 = *(const v8h*)&sKh[(16 + laneLo) * 8] * hmask;
            v8h u0 = *(const v8h*)&sKm[laneLo * 8]        * hmask;
            v8h u1 = *(const v8h*)&sKm[(16 + laneLo) * 8] * hmask;
            for (int i = 0; i < 8; i++) { bk0[i] = t0[i]; bk1[i] = t1[i]; bm0[i] = u0[i]; bm1[i] = u1[i]; }
        }
        v8f z = {};
        v8f Sh0 = wmma16(aQ, bk0, z);
        v8f Sh1 = wmma16(aQ, bk1, z);
        v8f Sm0 = wmma16(aQ, bm0, z);
        v8f Sm1 = wmma16(aQ, bm1, z);

        // Online softmax; row state sits in the natural D-layout mapping (VGPR v
        // <-> row v+8*laneHi); row reductions are 16-lane butterflies.
        auto update = [&](v8f S0, v8f S1, float* ms, float* ls, v8f* acc, _Float16* pbuf) {
            for (int v = 0; v < 8; v++) {
                float rmax = fmaxf(S0[v], S1[v]);
                for (int d = 1; d < 16; d <<= 1) rmax = fmaxf(rmax, __shfl_xor(rmax, d, 32));
                float mnew = fmaxf(ms[v], rmax);
                float corr = __expf(ms[v] - mnew);
                float p0 = __expf(S0[v] - mnew);
                float p1 = __expf(S1[v] - mnew);
                float rs = p0 + p1;
                for (int d = 1; d < 16; d <<= 1) rs += __shfl_xor(rs, d, 32);
                ls[v] = ls[v] * corr + rs;
                ms[v] = mnew;
                for (int t = 0; t < 4; t++) acc[t][v] *= corr;
                int row = v + 8 * laneHi;
                pbuf[row * 32 + laneLo]      = (_Float16)p0;
                pbuf[row * 32 + 16 + laneLo] = (_Float16)p1;
            }
        };
        update(Sh0, Sh1, mh,  lh, acch, &sP[wave][0][0]);
        update(Sm0, Sm1, mmx, lm, accm, &sP[wave][1][0]);
        __syncthreads();

        // P: D-layout -> A-layout via LDS round trip.
        v16h aPh, aPm;
        {
            int row = laneLo, base = 16 * laneHi;
            v8h lo = *(const v8h*)&sP[wave][0][row * 32 + base];
            v8h hi = *(const v8h*)&sP[wave][0][row * 32 + base + 8];
            for (int i = 0; i < 8; i++) { aPh[i] = lo[i]; aPh[8 + i] = hi[i]; }
            lo = *(const v8h*)&sP[wave][1][row * 32 + base];
            hi = *(const v8h*)&sP[wave][1][row * 32 + base + 8];
            for (int i = 0; i < 8; i++) { aPm[i] = lo[i]; aPm[8 + i] = hi[i]; }
        }
        // Accumulate O += P * V^T over the 4 channel tiles.
        for (int t = 0; t < 4; t++) {
            v16h bVh = *(const v16h*)&sVh[(t * 16 + laneLo) * 32 + 16 * laneHi];
            v16h bVm = *(const v16h*)&sVm[(t * 16 + laneLo) * 32 + 16 * laneHi];
            acch[t] = wmma16(aPh, bVh, acch[t]);
            accm[t] = wmma16(aPm, bVm, accm[t]);
        }
    }

    for (int v = 0; v < 8; v++) { lh[v] = 1.0f / lh[v]; lm[v] = 1.0f / lm[v]; }
    for (int t = 0; t < 4; t++)
        for (int v = 0; v < 8; v++) {
            int posr = v + 8 * laneHi;
            int c = t * 16 + laneLo;
            size_t rb = ((size_t)(b * NN + q0 + posr)) * 128;
            zcat[rb + c]      = (_Float16)(acch[t][v] * lh[v]);
            zcat[rb + 64 + c] = (_Float16)(accm[t][v] * lm[v]);
        }
}

// ---------------------------------------------------------------------------
// Epilogue: Z = wz*[Zh|Zm]+bz; gates O/G/I from [h|Z]; fused LSTM-style update.
// ---------------------------------------------------------------------------
__global__ __launch_bounds__(128) void gates_kernel(
    const _Float16* __restrict__ hT, const _Float16* __restrict__ zcat,
    const _Float16* __restrict__ wzP, const float* __restrict__ bz,
    const _Float16* __restrict__ woP, const float* __restrict__ bo,
    const _Float16* __restrict__ wgP, const float* __restrict__ bg,
    const _Float16* __restrict__ wiP, const float* __restrict__ bi,
    const float* __restrict__ Mprev, float* __restrict__ Hout, float* __restrict__ Mout) {
    __shared__ alignas(32) _Float16 sZ[4][16 * 64];
    const int tid = threadIdx.x, lane = tid & 31, wave = tid >> 5;
    const int b  = blockIdx.x / (NN / 64);
    const int n0 = (blockIdx.x % (NN / 64)) * 64 + wave * 16;
    const int laneHi = lane >> 4, laneLo = lane & 15;

    // B fragments from Zcat rows (cin = 0..127).
    v16h bZ[4];
    for (int kc = 0; kc < 4; kc++)
        bZ[kc] = *(const v16h*)&zcat[((size_t)(b * NN + n0 + laneLo)) * 128 + kc * 32 + 16 * laneHi];

    // Z = wz * Zcat + bz -> stash in LDS as [pos][cz] f16.
    for (int t = 0; t < 4; t++) {
        v8f acc = *(const v8f*)&bz[t * 16 + 8 * laneHi];
        for (int kc = 0; kc < 4; kc++) {
            v16h a = *(const v16h*)&wzP[((size_t)((t * 4 + kc) * 32 + lane)) * 16];
            acc = wmma16(a, bZ[kc], acc);
        }
        for (int v = 0; v < 8; v++) {
            int cz = t * 16 + v + 8 * laneHi;
            sZ[wave][laneLo * 64 + cz] = (_Float16)acc[v];
        }
    }
    __syncthreads();

    // Gate B fragments: cin 0..63 from hT, 64..127 from Z.
    v16h bG[4];
    for (int kc = 0; kc < 2; kc++)
        bG[kc] = *(const v16h*)&hT[((size_t)(b * NN + n0 + laneLo)) * CC + kc * 32 + 16 * laneHi];
    for (int kc = 2; kc < 4; kc++)
        bG[kc] = *(const v16h*)&sZ[wave][laneLo * 64 + (kc - 2) * 32 + 16 * laneHi];

    v8f accO[4], accG[4], accI[4];
    for (int t = 0; t < 4; t++) {
        v8f ao = *(const v8f*)&bo[t * 16 + 8 * laneHi];
        v8f ag = *(const v8f*)&bg[t * 16 + 8 * laneHi];
        v8f ai = *(const v8f*)&bi[t * 16 + 8 * laneHi];
        for (int kc = 0; kc < 4; kc++) {
            size_t wofs = ((size_t)((t * 4 + kc) * 32 + lane)) * 16;
            ao = wmma16(*(const v16h*)&woP[wofs], bG[kc], ao);
            ag = wmma16(*(const v16h*)&wgP[wofs], bG[kc], ag);
            ai = wmma16(*(const v16h*)&wiP[wofs], bG[kc], ai);
        }
        accO[t] = ao; accG[t] = ag; accI[t] = ai;
    }

    for (int t = 0; t < 4; t++)
        for (int v = 0; v < 8; v++) {
            int c = t * 16 + v + 8 * laneHi;
            size_t idx = ((size_t)(b * CC + c)) * NN + n0 + laneLo;   // coalesced in laneLo
            float mv = Mprev[idx];
            float O = 1.0f / (1.0f + __expf(-accO[t][v]));
            float G = tanhf(accG[t][v]);
            float I = 1.0f / (1.0f + __expf(-accI[t][v]));
            float Mt = (1.0f - I) * mv + I * G;
            Hout[idx] = O * Mt;
            Mout[idx] = Mt;
        }
}

// ---------------------------------------------------------------------------
extern "C" void kernel_launch(void* const* d_in, const int* in_sizes, int n_in,
                              void* d_out, int out_size, void* d_ws, size_t ws_size,
                              hipStream_t stream) {
    (void)in_sizes; (void)n_in; (void)out_size; (void)ws_size;
    const float* Hin   = (const float*)d_in[0];
    const float* Mprev = (const float*)d_in[1];
    const float* wq  = (const float*)d_in[2];  const float* bq  = (const float*)d_in[3];
    const float* wkh = (const float*)d_in[4];  const float* bkh = (const float*)d_in[5];
    const float* wvh = (const float*)d_in[6];  const float* bvh = (const float*)d_in[7];
    const float* wkm = (const float*)d_in[8];  const float* bkm = (const float*)d_in[9];
    const float* wvm = (const float*)d_in[10]; const float* bvm = (const float*)d_in[11];
    const float* wz  = (const float*)d_in[12]; const float* bz  = (const float*)d_in[13];
    const float* wo  = (const float*)d_in[14]; const float* bo  = (const float*)d_in[15];
    const float* wg  = (const float*)d_in[16]; const float* bg  = (const float*)d_in[17];
    const float* wi  = (const float*)d_in[18]; const float* bi  = (const float*)d_in[19];

    _Float16* p = (_Float16*)d_ws;
    _Float16* hT    = p; p += (size_t)BB * NN * CC;
    _Float16* mT    = p; p += (size_t)BB * NN * CC;
    _Float16* qrow  = p; p += (size_t)BB * NN * QKD;
    _Float16* khrow = p; p += (size_t)BB * NN * QKD;
    _Float16* kmrow = p; p += (size_t)BB * NN * QKD;
    _Float16* vh    = p; p += (size_t)BB * CC * NN;
    _Float16* vm    = p; p += (size_t)BB * CC * NN;
    _Float16* zc    = p; p += (size_t)BB * NN * 128;
    _Float16* wqP  = p; p += 1 * 2 * 32 * 16;
    _Float16* wkhP = p; p += 1 * 2 * 32 * 16;
    _Float16* wkmP = p; p += 1 * 2 * 32 * 16;
    _Float16* wvhP = p; p += 4 * 2 * 32 * 16;
    _Float16* wvmP = p; p += 4 * 2 * 32 * 16;
    _Float16* wzP  = p; p += 4 * 4 * 32 * 16;
    _Float16* woP  = p; p += 4 * 4 * 32 * 16;
    _Float16* wgP  = p; p += 4 * 4 * 32 * 16;
    _Float16* wiP  = p; p += 4 * 4 * 32 * 16;
    float* fb = (float*)p;                      // 32B-aligned (all counts above are even)
    float* bqP  = fb; fb += 16;
    float* bkhP = fb; fb += 16;
    float* bkmP = fb; fb += 16;

    const float s = 0.35355339059f;             // 1/sqrt(QK)
    pack_w<<<4,  256, 0, stream>>>(wq,  wqP,  QKD, CC,  1, 2, s);
    pack_w<<<4,  256, 0, stream>>>(wkh, wkhP, QKD, CC,  1, 2, 1.0f);
    pack_w<<<4,  256, 0, stream>>>(wkm, wkmP, QKD, CC,  1, 2, 1.0f);
    pack_w<<<16, 256, 0, stream>>>(wvh, wvhP, CC,  CC,  4, 2, 1.0f);
    pack_w<<<16, 256, 0, stream>>>(wvm, wvmP, CC,  CC,  4, 2, 1.0f);
    pack_w<<<32, 256, 0, stream>>>(wz,  wzP,  CC,  128, 4, 4, 1.0f);
    pack_w<<<32, 256, 0, stream>>>(wo,  woP,  CC,  128, 4, 4, 1.0f);
    pack_w<<<32, 256, 0, stream>>>(wg,  wgP,  CC,  128, 4, 4, 1.0f);
    pack_w<<<32, 256, 0, stream>>>(wi,  wiP,  CC,  128, 4, 4, 1.0f);
    pack_bias8<<<1, 16, 0, stream>>>(bq,  bqP,  s);
    pack_bias8<<<1, 16, 0, stream>>>(bkh, bkhP, 1.0f);
    pack_bias8<<<1, 16, 0, stream>>>(bkm, bkmP, 1.0f);

    transpose_f16<<<(BB * CC * NN) / 256, 256, 0, stream>>>(Hin, Mprev, hT, mT);

    dim3 grid(BB * (NN / 64));
    proj_kernel<<<grid, 128, 0, stream>>>(hT, mT, wqP, bqP, wkhP, bkhP, wkmP, bkmP,
                                          wvhP, bvh, wvmP, bvm,
                                          qrow, khrow, kmrow, vh, vm);
    attn_kernel<<<grid, 128, 0, stream>>>(qrow, khrow, kmrow, vh, vm, zc);
    gates_kernel<<<grid, 128, 0, stream>>>(hT, zc, wzP, bz, woP, bo, wgP, bg, wiP, bi,
                                           Mprev, (float*)d_out,
                                           (float*)d_out + (size_t)BB * CC * NN);
}
